// Spatial_Transformer_87316685127713
// MI455X (gfx1250) — matrix operand order repaired and checked by reading, not compiled, not run on hardware
//
#include <hip/hip_runtime.h>
#include <cstdint>

#define DIMX      512
#define HEADS     8
#define DIM_HEAD  64
#define MLP_DIM   2048
#define BATCH     8
#define SEQ       1024
#define TOKENS    (BATCH * SEQ)      // 8192
#define DEPTH     4
#define INNER     (HEADS * DIM_HEAD) // 512
#define QKV_N     (3 * INNER)        // 1536
#define LN_EPS    1e-5f
#define ATT_SCALE 0.125f             // 64^-0.5

#if defined(__has_builtin)
#if __has_builtin(__builtin_amdgcn_global_load_async_to_lds_b128) && \
    __has_builtin(__builtin_amdgcn_s_wait_asynccnt)
#define USE_ASYNC_COPY 1
#endif
#endif

typedef __attribute__((ext_vector_type(4)))  unsigned int u32x4;
typedef __attribute__((ext_vector_type(4)))  float        f32x4;
typedef __attribute__((ext_vector_type(16))) __bf16       v16bf;
typedef __attribute__((ext_vector_type(8)))  float        v8f;
typedef int v4i32 __attribute__((vector_size(16)));

__device__ __forceinline__ unsigned short f2bf(float f) {
  unsigned int u = __float_as_uint(f);
  u += 0x7fffu + ((u >> 16) & 1u);          // round-to-nearest-even
  return (unsigned short)(u >> 16);
}

__device__ __forceinline__ v8f v8f_zero() {
  v8f z;
#pragma unroll
  for (int i = 0; i < 8; ++i) z[i] = 0.f;
  return z;
}

// Build a 16xbf16 fragment from two 16-byte chunks (LDS or global).
__device__ __forceinline__ v16bf ld_frag(const unsigned short* p0,
                                         const unsigned short* p1) {
  v16bf r;
  ((u32x4*)&r)[0] = *(const u32x4*)p0;
  ((u32x4*)&r)[1] = *(const u32x4*)p1;
  return r;
}

__device__ __forceinline__ v8f wmma_bf16(v16bf a, v16bf b, v8f c) {
  return __builtin_amdgcn_wmma_f32_16x16x32_bf16(
      /*neg_a=*/false, a, /*neg_b=*/false, b,
      /*c_mod=*/(short)0, c, /*reuse_a=*/false, /*reuse_b=*/false);
}

#ifdef USE_ASYNC_COPY
// per-lane 16B global -> LDS copy via GLOBAL_LOAD_ASYNC_TO_LDS_B128 (ASYNCcnt)
__device__ __forceinline__ void async_copy16(const unsigned short* g,
                                             unsigned short* l) {
  __builtin_amdgcn_global_load_async_to_lds_b128(
      (__attribute__((address_space(1))) v4i32*)g,
      (__attribute__((address_space(3))) v4i32*)l, 0, 0);
}
#endif

// reduce over the 16 lanes of each half-wave (xor masks < 16 never cross halves)
__device__ __forceinline__ float hred_max(float v) {
  v = fmaxf(v, __shfl_xor(v, 1, 32));
  v = fmaxf(v, __shfl_xor(v, 2, 32));
  v = fmaxf(v, __shfl_xor(v, 4, 32));
  v = fmaxf(v, __shfl_xor(v, 8, 32));
  return v;
}
__device__ __forceinline__ float hred_add(float v) {
  v += __shfl_xor(v, 1, 32);
  v += __shfl_xor(v, 2, 32);
  v += __shfl_xor(v, 4, 32);
  v += __shfl_xor(v, 8, 32);
  return v;
}

// ---------------------------------------------------------------------------
// fp32 -> bf16 conversion (weights)
// ---------------------------------------------------------------------------
__global__ void cvt_f32_bf16(const float* __restrict__ in,
                             unsigned short* __restrict__ out, int n) {
  int i = blockIdx.x * blockDim.x + threadIdx.x;
  if (i < n) out[i] = f2bf(in[i]);
}

// ---------------------------------------------------------------------------
// LayerNorm (fp32 in) fused with bf16 output. One block per token, 128 thr.
// ---------------------------------------------------------------------------
__global__ __launch_bounds__(128)
void ln_bf16(const float* __restrict__ x, const float* __restrict__ g,
             const float* __restrict__ bta, unsigned short* __restrict__ out) {
  __shared__ float rs[128], rq[128];
  const int t = blockIdx.x;
  const int tid = threadIdx.x;
  const float* xp = x + (size_t)t * DIMX;
  f32x4 v = *(const f32x4*)(xp + tid * 4);
  float s = v[0] + v[1] + v[2] + v[3];
  float q = v[0]*v[0] + v[1]*v[1] + v[2]*v[2] + v[3]*v[3];
  rs[tid] = s; rq[tid] = q;
  __syncthreads();
  for (int off = 64; off > 0; off >>= 1) {
    if (tid < off) { rs[tid] += rs[tid+off]; rq[tid] += rq[tid+off]; }
    __syncthreads();
  }
  const float mu   = rs[0] * (1.f / DIMX);
  const float var  = rq[0] * (1.f / DIMX) - mu * mu;
  const float rstd = rsqrtf(var + LN_EPS);
  unsigned short* op = out + (size_t)t * DIMX;
#pragma unroll
  for (int i = 0; i < 4; ++i) {
    const int c = tid * 4 + i;
    op[c] = f2bf((v[i] - mu) * rstd * g[c] + bta[c]);
  }
}

// ---------------------------------------------------------------------------
// Tiled bf16 GEMM, C = A[MxK] * B[KxN], 128x128 block tile, 8 waves,
// double-buffered LDS (one barrier per K-step, fetch overlapped with WMMA).
// EPI 0: fp32 out = acc + bias + res      (residual add)
// EPI 1: bf16 out = leakyrelu(acc + bias) (FFN1)
// EPI 2: bf16 out = acc                   (QKV)
// ---------------------------------------------------------------------------
template <int EPI>
__global__ __launch_bounds__(256)
void gemm_bf16(const unsigned short* __restrict__ A,
               const unsigned short* __restrict__ Bw,
               const float* __restrict__ bias,
               const float* __restrict__ res,
               void* __restrict__ outp, int M, int N, int K) {
  __shared__ __align__(16) unsigned short As[2][128 * 40]; // [m][k], ld=40
  __shared__ __align__(16) unsigned short Bt[2][128 * 40]; // [n][k], ld=40

  const int tid  = threadIdx.x;
  const int lane = tid & 31;
  const int wid  = tid >> 5;
  const int wy = wid >> 2, wx = wid & 3;     // 2 x 4 wave grid
  const int nl = lane & 15, hh = lane >> 4;

  const int m0 = blockIdx.y * 128;
  const int n0 = blockIdx.x * 128;

  v8f acc[4][2];
#pragma unroll
  for (int i = 0; i < 4; ++i)
#pragma unroll
    for (int j = 0; j < 2; ++j) acc[i][j] = v8f_zero();

  const int arow  = tid >> 1, akseg = (tid & 1) * 16;
  const int bkrow = tid >> 3, bnseg = (tid & 7) * 16;
  const unsigned short* aSrc = A  + (size_t)(m0 + arow) * K + akseg;
  const unsigned short* bSrc = Bw + (size_t)bkrow * N + n0 + bnseg;
  const int aDst = arow * 40 + akseg;

  // ---- prologue: stage tile 0 into buffer 0 ----
#ifdef USE_ASYNC_COPY
  async_copy16(aSrc,     &As[0][aDst]);
  async_copy16(aSrc + 8, &As[0][aDst + 8]);
#else
  {
    u32x4 ra0 = *(const u32x4*)aSrc;
    u32x4 ra1 = *(const u32x4*)(aSrc + 8);
    *(u32x4*)&As[0][aDst]     = ra0;
    *(u32x4*)&As[0][aDst + 8] = ra1;
  }
#endif
  {
    u32x4 w0 = *(const u32x4*)bSrc;
    u32x4 w1 = *(const u32x4*)(bSrc + 8);
#pragma unroll
    for (int i = 0; i < 4; ++i) {
      Bt[0][(bnseg + 2*i    ) * 40 + bkrow] = (unsigned short)(w0[i] & 0xffffu);
      Bt[0][(bnseg + 2*i + 1) * 40 + bkrow] = (unsigned short)(w0[i] >> 16);
      Bt[0][(bnseg + 2*i + 8) * 40 + bkrow] = (unsigned short)(w1[i] & 0xffffu);
      Bt[0][(bnseg + 2*i + 9) * 40 + bkrow] = (unsigned short)(w1[i] >> 16);
    }
  }

  const int NT = K >> 5;
  for (int t = 0; t < NT; ++t) {
    const int p = t & 1;
    const bool nxt = (t + 1 < NT);
    u32x4 rb0, rb1;
#ifndef USE_ASYNC_COPY
    u32x4 ra0, ra1;
#endif
    if (nxt) { // issue next-tile global loads early (latency overlap)
      const unsigned short* bs = bSrc + (size_t)(t + 1) * 32 * N;
      rb0 = *(const u32x4*)bs;
      rb1 = *(const u32x4*)(bs + 8);
#ifndef USE_ASYNC_COPY
      const unsigned short* as = aSrc + (t + 1) * 32;
      ra0 = *(const u32x4*)as;
      ra1 = *(const u32x4*)(as + 8);
#endif
    }
#ifdef USE_ASYNC_COPY
    __builtin_amdgcn_s_wait_asynccnt(0); // tile t's async copies landed
#endif
    __syncthreads(); // readers of buffer p^1 (iter t-1) are done; buf p valid

    if (nxt) { // stage tile t+1 into buffer p^1 (now safe)
#ifdef USE_ASYNC_COPY
      const unsigned short* as = aSrc + (t + 1) * 32;
      async_copy16(as,     &As[p ^ 1][aDst]);
      async_copy16(as + 8, &As[p ^ 1][aDst + 8]);
#else
      *(u32x4*)&As[p ^ 1][aDst]     = ra0;
      *(u32x4*)&As[p ^ 1][aDst + 8] = ra1;
#endif
      unsigned short* btn = &Bt[p ^ 1][0];
#pragma unroll
      for (int i = 0; i < 4; ++i) {
        btn[(bnseg + 2*i    ) * 40 + bkrow] = (unsigned short)(rb0[i] & 0xffffu);
        btn[(bnseg + 2*i + 1) * 40 + bkrow] = (unsigned short)(rb0[i] >> 16);
        btn[(bnseg + 2*i + 8) * 40 + bkrow] = (unsigned short)(rb1[i] & 0xffffu);
        btn[(bnseg + 2*i + 9) * 40 + bkrow] = (unsigned short)(rb1[i] >> 16);
      }
    }

    // ---- compute from buffer p ----
    v16bf bf[2];
#pragma unroll
    for (int ns = 0; ns < 2; ++ns) {
      const unsigned short* bp = &Bt[p][(wx*32 + ns*16 + nl) * 40 + 8*hh];
      bf[ns] = ld_frag(bp, bp + 16);              // K[8h..), K[16+8h..)
    }
#pragma unroll
    for (int ms = 0; ms < 4; ++ms) {
      const unsigned short* ap = &As[p][(wy*64 + ms*16 + nl) * 40 + 8*hh];
      v16bf af = ld_frag(ap, ap + 16);            // K[8h..), K[16+8h..)
#pragma unroll
      for (int ns = 0; ns < 2; ++ns)
        acc[ms][ns] = wmma_bf16(af, bf[ns], acc[ms][ns]);
    }
  }

  const int orow0 = m0 + wy * 64;
  const int ocol0 = n0 + wx * 32;
#pragma unroll
  for (int ms = 0; ms < 4; ++ms) {
#pragma unroll
    for (int ns = 0; ns < 2; ++ns) {
      const int col = ocol0 + ns*16 + nl;
      float bv = 0.f;
      if constexpr (EPI != 2) bv = bias[col];
#pragma unroll
      for (int r = 0; r < 8; ++r) {
        const int row = orow0 + ms*16 + r + 8*hh;
        const size_t idx = (size_t)row * N + col;
        float v = acc[ms][ns][r];
        if constexpr (EPI == 0) {
          ((float*)outp)[idx] = v + bv + res[idx];
        } else if constexpr (EPI == 1) {
          v += bv;
          v = (v >= 0.f) ? v : 0.01f * v;
          ((unsigned short*)outp)[idx] = f2bf(v);
        } else {
          ((unsigned short*)outp)[idx] = f2bf(v);
        }
      }
    }
  }
}

// ---------------------------------------------------------------------------
// Flash attention: one block per (b, h, 64 query rows); 4 waves x 16 rows.
// qkv: bf16 [TOKENS, 1536] (q | k | v); out: bf16 [TOKENS, 512]
// ---------------------------------------------------------------------------
__global__ __launch_bounds__(128)
void attention_kernel(const unsigned short* __restrict__ qkv,
                      unsigned short* __restrict__ attn_out) {
  __shared__ __align__(16) unsigned short Ks[32 * 72];      // [key][d], ld=72
  __shared__ __align__(16) unsigned short VT[64 * 40];      // [d][key], ld=40
  __shared__ __align__(16) unsigned short Pw[4 * 16 * 40];  // per-wave P tile

  const int tid  = threadIdx.x;
  const int lane = tid & 31;
  const int w    = tid >> 5;
  const int nl = lane & 15, hh = lane >> 4;

  const int qb = blockIdx.x;            // query block (64 rows)
  const int h  = blockIdx.y;
  const int b  = blockIdx.z;

  const size_t tok0 = (size_t)b * SEQ;
  const int qrow0 = qb * 64 + w * 16;

  v16bf qf[2];
  {
    const unsigned short* qrow =
        qkv + (tok0 + qrow0 + nl) * QKV_N + h * DIM_HEAD;
#pragma unroll
    for (int f = 0; f < 2; ++f)
      qf[f] = ld_frag(qrow + f*32 + 8*hh, qrow + f*32 + 16 + 8*hh);
  }

  v8f o[4];
#pragma unroll
  for (int f = 0; f < 4; ++f) o[f] = v8f_zero();
  float Mr[8], Lr[8];
#pragma unroll
  for (int r = 0; r < 8; ++r) { Mr[r] = -1e30f; Lr[r] = 0.f; }

  const unsigned short* Kbase = qkv + tok0 * QKV_N + INNER     + h * DIM_HEAD;
  const unsigned short* Vbase = qkv + tok0 * QKV_N + 2 * INNER + h * DIM_HEAD;

  const int skk = tid >> 2;            // staging: key row 0..31
  const int sds = (tid & 3) * 16;      // staging: d segment
  unsigned short* myP = &Pw[w * 16 * 40];

  for (int j = 0; j < SEQ; j += 32) {
    __syncthreads();
    { // stage 32 K rows (row-major copy) -- async path if available
      const unsigned short* src = Kbase + (size_t)(j + skk) * QKV_N + sds;
#ifdef USE_ASYNC_COPY
      async_copy16(src,     &Ks[skk * 72 + sds]);
      async_copy16(src + 8, &Ks[skk * 72 + sds + 8]);
#else
      *(u32x4*)&Ks[skk * 72 + sds]     = *(const u32x4*)src;
      *(u32x4*)&Ks[skk * 72 + sds + 8] = *(const u32x4*)(src + 8);
#endif
    }
    { // stage 32 V rows transposed -> VT[d][key]
      const unsigned short* src = Vbase + (size_t)(j + skk) * QKV_N + sds;
      u32x4 v0 = *(const u32x4*)src;
      u32x4 v1 = *(const u32x4*)(src + 8);
#pragma unroll
      for (int i = 0; i < 4; ++i) {
        VT[(sds + 2*i    ) * 40 + skk] = (unsigned short)(v0[i] & 0xffffu);
        VT[(sds + 2*i + 1) * 40 + skk] = (unsigned short)(v0[i] >> 16);
        VT[(sds + 2*i + 8) * 40 + skk] = (unsigned short)(v1[i] & 0xffffu);
        VT[(sds + 2*i + 9) * 40 + skk] = (unsigned short)(v1[i] >> 16);
      }
    }
#ifdef USE_ASYNC_COPY
    __builtin_amdgcn_s_wait_asynccnt(0);
#endif
    __syncthreads();

    // S = Q * K^T for two 16-key tiles, accumulate over d (two k-steps)
    v8f s[2];
#pragma unroll
    for (int c = 0; c < 2; ++c) {
      s[c] = v8f_zero();
#pragma unroll
      for (int f = 0; f < 2; ++f) {
        const unsigned short* kp = &Ks[(c*16 + nl) * 72 + f*32 + 8*hh];
        v16bf kb = ld_frag(kp, kp + 16);
        s[c] = wmma_bf16(qf[f], kb, s[c]);
      }
#pragma unroll
      for (int r = 0; r < 8; ++r) s[c][r] *= ATT_SCALE;
    }

    // online softmax per row (row r+8*hh lives in the 16 lanes of a half)
#pragma unroll
    for (int r = 0; r < 8; ++r) {
      float mx = hred_max(fmaxf(s[0][r], s[1][r]));
      float Mn = fmaxf(Mr[r], mx);
      float fac = __expf(Mr[r] - Mn);
      Mr[r] = Mn;
      float p0 = __expf(s[0][r] - Mn);
      float p1 = __expf(s[1][r] - Mn);
      s[0][r] = p0; s[1][r] = p1;
      float rs = hred_add(p0 + p1);
      Lr[r] = Lr[r] * fac + rs;
#pragma unroll
      for (int f = 0; f < 4; ++f) o[f][r] *= fac;
    }

    // P (16x32) C-layout -> A-fragment via per-wave LDS round trip
#pragma unroll
    for (int c = 0; c < 2; ++c)
#pragma unroll
      for (int r = 0; r < 8; ++r)
        myP[(r + 8*hh) * 40 + c*16 + nl] = f2bf(s[c][r]);

    const unsigned short* pp = &myP[nl * 40 + 8*hh];
    v16bf pf = ld_frag(pp, pp + 16);

    // O += P * V  (four 16-wide d sub-tiles)
#pragma unroll
    for (int f = 0; f < 4; ++f) {
      const unsigned short* vp = &VT[(f*16 + nl) * 40 + 8*hh];
      v16bf vb = ld_frag(vp, vp + 16);
      o[f] = wmma_bf16(pf, vb, o[f]);
    }
  }

#pragma unroll
  for (int r = 0; r < 8; ++r) Lr[r] = 1.f / Lr[r];
  unsigned short* obase = attn_out + (tok0 + qrow0) * INNER + h * DIM_HEAD;
#pragma unroll
  for (int f = 0; f < 4; ++f)
#pragma unroll
    for (int r = 0; r < 8; ++r)
      obase[(size_t)(r + 8*hh) * INNER + f*16 + nl] = f2bf(o[f][r] * Lr[r]);
}

// ---------------------------------------------------------------------------
extern "C" void kernel_launch(void* const* d_in, const int*, int,
                              void* d_out, int, void* d_ws, size_t,
                              hipStream_t stream) {
  const float* x_in   = (const float*)d_in[0];
  const float* ln1_g  = (const float*)d_in[1];
  const float* ln1_b  = (const float*)d_in[2];
  const float* qkv_w  = (const float*)d_in[3];
  const float* out_w  = (const float*)d_in[4];
  const float* out_b  = (const float*)d_in[5];
  const float* ln2_g  = (const float*)d_in[6];
  const float* ln2_b  = (const float*)d_in[7];
  const float* ffn_w1 = (const float*)d_in[8];
  const float* ffn_b1 = (const float*)d_in[9];
  const float* ffn_w2 = (const float*)d_in[10];
  const float* ffn_b2 = (const float*)d_in[11];

  char* ws = (char*)d_ws;
  float*          x    = (float*)(ws);                                     // 16 MB
  unsigned short* xh   = (unsigned short*)(ws + (16u << 20));              //  8 MB
  unsigned short* wbf  = (unsigned short*)(ws + (16u << 20) + (8u << 20)); //  2 MB
  unsigned short* qkvb = (unsigned short*)(ws + (16u << 20) + (8u << 20) + (2u << 20));
  unsigned short* attn = qkvb + (size_t)TOKENS * QKV_N;                    //  8 MB
  unsigned short* hbuf = qkvb; // FFN hidden reuses qkv+attn region (disjoint lifetime)

  (void)hipMemcpyAsync(x, x_in, (size_t)TOKENS * DIMX * sizeof(float),
                       hipMemcpyDeviceToDevice, stream);

  for (int l = 0; l < DEPTH; ++l) {
    ln_bf16<<<TOKENS, 128, 0, stream>>>(x, ln1_g + l*DIMX, ln1_b + l*DIMX, xh);

    cvt_f32_bf16<<<(DIMX*QKV_N + 255)/256, 256, 0, stream>>>(
        qkv_w + (size_t)l*DIMX*QKV_N, wbf, DIMX*QKV_N);
    gemm_bf16<2><<<dim3(QKV_N/128, TOKENS/128), 256, 0, stream>>>(
        xh, wbf, nullptr, nullptr, qkvb, TOKENS, QKV_N, DIMX);

    attention_kernel<<<dim3(SEQ/64, HEADS, BATCH), 128, 0, stream>>>(qkvb, attn);

    cvt_f32_bf16<<<(INNER*DIMX + 255)/256, 256, 0, stream>>>(
        out_w + (size_t)l*INNER*DIMX, wbf, INNER*DIMX);
    gemm_bf16<0><<<dim3(DIMX/128, TOKENS/128), 256, 0, stream>>>(
        attn, wbf, out_b + l*DIMX, x, x, TOKENS, DIMX, INNER);

    ln_bf16<<<TOKENS, 128, 0, stream>>>(x, ln2_g + l*DIMX, ln2_b + l*DIMX, xh);

    cvt_f32_bf16<<<(DIMX*MLP_DIM + 255)/256, 256, 0, stream>>>(
        ffn_w1 + (size_t)l*DIMX*MLP_DIM, wbf, DIMX*MLP_DIM);
    gemm_bf16<1><<<dim3(MLP_DIM/128, TOKENS/128), 256, 0, stream>>>(
        xh, wbf, ffn_b1 + l*MLP_DIM, nullptr, hbuf, TOKENS, MLP_DIM, DIMX);

    cvt_f32_bf16<<<(MLP_DIM*DIMX + 255)/256, 256, 0, stream>>>(
        ffn_w2 + (size_t)l*MLP_DIM*DIMX, wbf, MLP_DIM*DIMX);
    gemm_bf16<0><<<dim3(DIMX/128, TOKENS/128), 256, 0, stream>>>(
        hbuf, wbf, ffn_b2 + l*DIMX, x, x, TOKENS, DIMX, MLP_DIM);
  }

  (void)hipMemcpyAsync(d_out, x, (size_t)TOKENS * DIMX * sizeof(float),
                       hipMemcpyDeviceToDevice, stream);
}
